// QuantLinear_31834297598371
// MI455X (gfx1250) — matrix operand order repaired and checked by reading, not compiled, not run on hardware
//
#include <hip/hip_runtime.h>

typedef _Float16 h16;
typedef __attribute__((ext_vector_type(16))) _Float16 v16h;
typedef __attribute__((ext_vector_type(8)))  _Float16 v8h;
typedef __attribute__((ext_vector_type(2)))  _Float16 v2h;
typedef __attribute__((ext_vector_type(8)))  float    v8f;
typedef __attribute__((ext_vector_type(4)))  int      v4i;
typedef __attribute__((ext_vector_type(2)))  int      v2i;
typedef __attribute__((ext_vector_type(4)))  unsigned v4u;

#define K_DIM 4096
#define N_DIM 4096
#define BM 256
#define BN 128
#define BK 32
#define LDA 40   // padded LDS row stride (halves), 80 B: 16B-aligned, conflict-free
#define LDB 40

#if defined(__AMDGCN__) && __has_builtin(__builtin_amdgcn_global_load_async_to_lds_b128)
#define USE_ASYNC_LDS 1
typedef __attribute__((address_space(1))) v4i g_v4i;
typedef __attribute__((address_space(3))) v4i l_v4i;
template <int N> __device__ __forceinline__ void wait_asynccnt() {
#if __has_builtin(__builtin_amdgcn_s_wait_asynccnt)
    __builtin_amdgcn_s_wait_asynccnt(N);
#else
    asm volatile("s_wait_asynccnt %0" ::"n"(N));
#endif
}
template <int OFF>
__device__ __forceinline__ void async_b128(const h16* g, h16* l) {
    __builtin_amdgcn_global_load_async_to_lds_b128((g_v4i*)g, (l_v4i*)l, OFF, 0);
}
#else
#define USE_ASYNC_LDS 0
#endif

__global__ __launch_bounds__(256)
void qgemm_wmma_f16(const h16* __restrict__ X, const int* __restrict__ QW,
                    const int* __restrict__ QZ, const h16* __restrict__ SC,
                    const h16* __restrict__ BIAS, h16* __restrict__ OUT)
{
    __shared__ h16 sX[2][BM * LDA];   // 2 x 20480 B
    __shared__ h16 sW[BN * LDB];      // 10240 B  (N-major: sW[n][k])

    const int t    = threadIdx.x;
    const int lane = t & 31;
    const int wid  = t >> 5;       // 8 waves
    const int wm   = wid & 3;      // 4 waves along M (64 rows each)
    const int wn   = wid >> 2;     // 2 waves along N (64 cols each)

    const int m0 = blockIdx.x * BM;
    const int n0 = blockIdx.y * BN;

    // X tile: 256 rows x 32 halves; 1 thread/row, 64 B = 4 x b128.
    // W tile: 4 int32 k-rows x 128 n; 2 int32 (one krow, 2 n) per thread.
    const int krow = t >> 6;              // 0..3
    const int n2   = (t & 63) * 2;        // 0..126

    v8f acc[4][4];
#pragma unroll
    for (int i = 0; i < 4; ++i)
#pragma unroll
        for (int j = 0; j < 4; ++j)
            acc[i][j] = (v8f)(0.0f);

    // W staging registers
    v2i qw;
    v2h sc2;
    int zq;
#if !USE_ASYNC_LDS
    v4u xr[4];
#endif

    const h16* xbase = X + (size_t)(m0 + t) * K_DIM;

    auto stage_x = [&](int kt) {
        const h16* xp = xbase + kt * BK;
        h16*       lp = &sX[kt & 1][t * LDA];
#if USE_ASYNC_LDS
        async_b128<0>(xp, lp);
        async_b128<16>(xp, lp);
        async_b128<32>(xp, lp);
        async_b128<48>(xp, lp);
#else
#pragma unroll
        for (int i = 0; i < 4; ++i)
            xr[i] = *(const v4u*)(xp + i * 8);
        (void)lp;
#endif
    };

    auto commit_x = [&](int kt) {
#if !USE_ASYNC_LDS
        h16* lp = &sX[kt & 1][t * LDA];
#pragma unroll
        for (int i = 0; i < 4; ++i)
            *(v4u*)(lp + i * 8) = xr[i];
#else
        (void)kt;
#endif
    };

    auto load_w = [&](int kt) {
        const int k0 = kt * BK;
        qw = *(const v2i*)(QW + (size_t)((k0 >> 3) + krow) * N_DIM + n0 + n2);
        const int g = k0 >> 7;                 // quant group (GROUPSIZE=128)
        zq  = QZ[(size_t)g * (N_DIM / 8) + ((n0 + n2) >> 3)];
        sc2 = *(const v2h*)(SC + (size_t)g * N_DIM + n0 + n2);
    };

    // W4 -> fp16 via exponent-magic: bitcast(0x6400|nib) == 1024+nib, so
    // w = pk_fma(magic_pair, {s,s}, {-s*(1024+z)}) dequantizes 2 elems/op.
    auto store_w = [&]() {
#pragma unroll
        for (int j = 0; j < 2; ++j) {
            const int   n  = n2 + j;
            const int   z  = (zq >> (4 * (n & 7))) & 0xF;
            const h16   s  = sc2[j];
            const h16   bh = (h16)(-(float)s * (float)(1024 + z));
            const v2h   S  = {s, s};
            const v2h   B  = {bh, bh};
            const unsigned q = (unsigned)qw[j];
            v8h w;
#pragma unroll
            for (int p = 0; p < 4; ++p) {
                const unsigned r = q >> (8 * p);
                const unsigned m = (r & 0xFu) | ((r << 12) & 0xF0000u) | 0x64006400u;
                const v2h hv  = __builtin_bit_cast(v2h, m);
                const v2h res = __builtin_elementwise_fma(hv, S, B);
                w[2 * p]     = res[0];
                w[2 * p + 1] = res[1];
            }
            *(v8h*)(sW + n * LDB + krow * 8) = w;
        }
    };

    const int NK = K_DIM / BK;   // 128 K-tiles
    stage_x(0);
    load_w(0);

    for (int kt = 0; kt < NK; ++kt) {
        __syncthreads();              // prev compute done: W LDS + X buf[(kt+1)&1] free
        commit_x(kt);
        store_w();
        if (kt + 1 < NK) {
            stage_x(kt + 1);          // overlap next tile's loads with compute
            load_w(kt + 1);
#if USE_ASYNC_LDS
            wait_asynccnt<4>();       // tile kt's 4 asyncs done (in-order completion)
#endif
        } else {
#if USE_ASYNC_LDS
            wait_asynccnt<0>();
#endif
        }
        __syncthreads();              // whole tile visible to all waves

        const h16* xt = sX[kt & 1];

        // A frags: lane row M=lane%16; halves 0..7 <- K=b+0..7,
        // 8..15 <- K=b+16..23, b = (lane>=16 ? 8 : 0).
        const int akb = (lane >> 4) << 3;
        v16h a[4];
#pragma unroll
        for (int tm = 0; tm < 4; ++tm) {
            const h16* ap = xt + (wm * 64 + tm * 16 + (lane & 15)) * LDA + akb;
            v8h lo = *(const v8h*)ap;
            v8h hi = *(const v8h*)(ap + 16);
            a[tm] = __builtin_shufflevector(lo, hi,
                    0,1,2,3,4,5,6,7,8,9,10,11,12,13,14,15);
        }
        // B frags: lane col N=lane%16; halves 0..7 <- K=b+0..7,
        // 8..15 <- K=b+16..23, b = (lane>=16 ? 8 : 0).
        const int bkb = (lane >> 4) << 3;
#pragma unroll
        for (int tn = 0; tn < 4; ++tn) {
            const h16* bp = sW + (wn * 64 + tn * 16 + (lane & 15)) * LDB + bkb;
            v8h lo = *(const v8h*)bp;
            v8h hi = *(const v8h*)(bp + 16);
            v16h b = __builtin_shufflevector(lo, hi,
                    0,1,2,3,4,5,6,7,8,9,10,11,12,13,14,15);
#pragma unroll
            for (int tm = 0; tm < 4; ++tm)
                acc[tm][tn] = __builtin_amdgcn_wmma_f32_16x16x32_f16(
                    false, a[tm], false, b, (short)0, acc[tm][tn],
                    false, false);
        }
    }

    // Epilogue: C layout — VGPR r: M = tile_m + (lane>=16?8:0) + r, N = lane%16.
#pragma unroll
    for (int tn = 0; tn < 4; ++tn) {
        const int n = n0 + wn * 64 + tn * 16 + (lane & 15);
        const float bf = (float)BIAS[n];
#pragma unroll
        for (int tm = 0; tm < 4; ++tm) {
            const int mb = m0 + wm * 64 + tm * 16 + ((lane >> 4) << 3);
            v8f c = acc[tm][tn];
#pragma unroll
            for (int r = 0; r < 8; ++r)
                OUT[(size_t)(mb + r) * N_DIM + n] = (h16)(c[r] + bf);
        }
    }
}

extern "C" void kernel_launch(void* const* d_in, const int* in_sizes, int n_in,
                              void* d_out, int out_size, void* d_ws, size_t ws_size,
                              hipStream_t stream) {
    const h16* x    = (const h16*)d_in[0];
    const int* qw   = (const int*)d_in[1];
    const int* qz   = (const int*)d_in[2];
    const h16* sc   = (const h16*)d_in[3];
    const h16* bias = (const h16*)d_in[4];
    h16* out        = (h16*)d_out;

    const int M = in_sizes[0] / K_DIM;            // 4*2048 = 8192
    dim3 grid(M / BM, N_DIM / BN);                // 32 x 32 workgroups
    qgemm_wmma_f16<<<grid, 256, 0, stream>>>(x, qw, qz, sc, bias, out);
}
